// MaskedTransformer_7095285973110
// MI455X (gfx1250) — matrix operand order repaired and checked by reading, not compiled, not run on hardware
//
#include <hip/hip_runtime.h>

// ---------------------------------------------------------------------------
// MaskedTransformer forward for gfx1250 (MI455X), bf16 WMMA + fp32 accumulate
// Double-buffered LDS; async global->LDS copies when the toolchain has them.
// ---------------------------------------------------------------------------

#define DEPTH 3
#define HEADS 8
#define DIM 512
#define MLP_DIM 4096
#define SEQ 2560
#define BATCH 2
#define NTOK (BATCH * SEQ)          // 5120
#define HD 64
#define SCALE_ATTN 0.125f           // (512/8)^-0.5

#if defined(__gfx1250__) && __has_builtin(__builtin_amdgcn_global_load_async_to_lds_b128) && __has_builtin(__builtin_amdgcn_s_wait_asynccnt)
#define USE_ASYNC 1
#else
#define USE_ASYNC 0
#endif

typedef float v8f __attribute__((ext_vector_type(8)));
typedef float v4f __attribute__((ext_vector_type(4)));
typedef __bf16 v16bf __attribute__((ext_vector_type(16)));
typedef unsigned int v4u __attribute__((ext_vector_type(4)));
typedef unsigned short v4us __attribute__((ext_vector_type(4)));

// types for the async global->LDS builtin (param 1 is AS1 int-vec16 pointer)
typedef int v4i_vs __attribute__((vector_size(16)));
typedef __attribute__((address_space(1))) v4i_vs* as1_v4i_p;
typedef __attribute__((address_space(3))) v4i_vs* as3_v4i_p;

union FragAB { v4u u[2]; v16bf v; };

__device__ __forceinline__ unsigned short f32_to_bf16(float f) {
  unsigned int u = __float_as_uint(f);
  u += 0x7FFFu + ((u >> 16) & 1u);        // round-to-nearest-even
  return (unsigned short)(u >> 16);
}

__device__ __forceinline__ float gelu_f(float x) {
  float x3 = x * x * x;
  return 0.5f * x * (1.0f + tanhf(0.7978845608028654f * (x + 0.044715f * x3)));
}

// One K-step of WMMA work for a wave: 4 A-frags x 2 B-frags -> 8 wmma.
// Fragment addressing follows the ISA 16-bit A/B VGPR layouts exactly.
__device__ __forceinline__ void wmma_tiles(const unsigned short* sAb,
                                           const unsigned short* sBb,
                                           int wm, int wn, int hi, int ln,
                                           v8f (&acc)[4][2]) {
  FragAB af[4], bf[2];
#pragma unroll
  for (int tm = 0; tm < 4; ++tm) {
    const unsigned short* p = sAb + (wm * 64 + tm * 16 + ln) * 40;
    af[tm].u[0] = *(const v4u*)(p + hi * 8);         // K = hi*8 .. +7
    af[tm].u[1] = *(const v4u*)(p + 16 + hi * 8);    // K = 16+hi*8 .. +7
  }
#pragma unroll
  for (int tn = 0; tn < 2; ++tn) {
    const unsigned short* p = sBb + (wn * 32 + tn * 16 + ln) * 40;
    bf[tn].u[0] = *(const v4u*)(p + hi * 8);         // K = hi*8 .. +7
    bf[tn].u[1] = *(const v4u*)(p + 16 + hi * 8);    // K = 16+hi*8 .. +7
  }
#pragma unroll
  for (int tm = 0; tm < 4; ++tm)
#pragma unroll
    for (int tn = 0; tn < 2; ++tn)
      acc[tm][tn] = __builtin_amdgcn_wmma_f32_16x16x32_bf16(
          false, af[tm].v, false, bf[tn].v, (short)0, acc[tm][tn], false, false);
}

// ---------------------------------------------------------------------------
// Generic batched GEMM:  C[m,n] = sum_k A[m,k] * Bt[n,k]
//   A: bf16 (or fp32 converted on stage when AF32), row-major, leading dim lda
//   Bt: bf16 row-major [N,K], leading dim ldb
// Block tile 128x128, K-step 32, 256 threads = 8 waves (2x4), wave tile 64x32.
// EPI: 0 = bf16 ; 1 = bf16 gelu(x+bias) ; 2 = f32 x*scale ; 3 = f32 res+x+bias
// GUARD: bounds-check M/N (only needed for the attn*V GEMM, N=64)
// ---------------------------------------------------------------------------
template <int EPI, bool AF32, bool GUARD>
__global__ __launch_bounds__(256) void gemm_k(
    const void* __restrict__ A, long long aBase, int lda, long long aSb, long long aSh,
    const unsigned short* __restrict__ Bt, long long bBase, int ldb, long long bSb, long long bSh,
    void* __restrict__ C, long long cBase, int ldc, long long cSb, long long cSh,
    const float* __restrict__ bias, const float* __restrict__ res,
    float scale, int M, int N, int K, int hdiv) {
  __shared__ __align__(16) unsigned short sA[2][128 * 40];
  __shared__ __align__(16) unsigned short sB[2][128 * 40];

  const int tid = threadIdx.x;
  const int z = blockIdx.z;
  const long long zb = z / hdiv, zh = z % hdiv;

  const unsigned short* Ab = (const unsigned short*)A + aBase + zb * aSb + zh * aSh;
  const float* Af = (const float*)A + aBase + zb * aSb + zh * aSh;
  const unsigned short* Bp = Bt + bBase + zb * bSb + zh * bSh;
  const long long cOff = cBase + zb * cSb + zh * cSh;

  const int bm0 = blockIdx.y * 128;
  const int bn0 = blockIdx.x * 128;

  const int w = tid >> 5;
  const int wm = w >> 2, wn = w & 3;          // 2 x 4 wave grid
  const int l = tid & 31;
  const int hi = l >> 4, ln = l & 15;

  const int rT = tid >> 2;                    // 0..63  (bf16 staging row)
  const int cT = (tid & 3) * 8;               // bf16 8-elem chunk
  const int rF = tid >> 3;                    // 0..31  (f32 staging row)
  const int cF = (tid & 7) * 4;               // f32 4-elem chunk

  const int nk = K >> 5;

  v8f zero = {0.f, 0.f, 0.f, 0.f, 0.f, 0.f, 0.f, 0.f};
  v8f acc[4][2];
#pragma unroll
  for (int i = 0; i < 4; ++i)
#pragma unroll
    for (int j = 0; j < 2; ++j) acc[i][j] = zero;

#if USE_ASYNC
  if constexpr (!AF32 && !GUARD) {
    // ---- fully async path: global -> LDS DMA, double-buffered ----
    auto issue = [&](int buf, int k0) {
#pragma unroll
      for (int it = 0; it < 2; ++it) {
        const unsigned short* g = Ab + (long long)(bm0 + rT + it * 64) * lda + k0 + cT;
        __builtin_amdgcn_global_load_async_to_lds_b128(
            (as1_v4i_p)(void*)g,
            (as3_v4i_p)(void*)&sA[buf][(rT + it * 64) * 40 + cT],
            0, 0);
      }
#pragma unroll
      for (int it = 0; it < 2; ++it) {
        const unsigned short* g = Bp + (long long)(bn0 + rT + it * 64) * ldb + k0 + cT;
        __builtin_amdgcn_global_load_async_to_lds_b128(
            (as1_v4i_p)(void*)g,
            (as3_v4i_p)(void*)&sB[buf][(rT + it * 64) * 40 + cT],
            0, 0);
      }
    };
    issue(0, 0);
    for (int ki = 0; ki < nk; ++ki) {
      const int cur = ki & 1;
      __builtin_amdgcn_s_wait_asynccnt(0);  // this wave's fills of buf[cur] done
      __syncthreads();                      // everyone's fills of buf[cur] done
      if (ki + 1 < nk) issue(cur ^ 1, (ki + 1) << 5);
      wmma_tiles(sA[cur], sB[cur], wm, wn, hi, ln, acc);
    }
  } else
#endif
  {
    // ---- register-staged path (fp32 conversion and/or ragged N), dbl-buffered
    v4f fa[4];
    v4u ua[2], ub[2];
    auto loadRegs = [&](int k0) {
      if (AF32) {
#pragma unroll
        for (int it = 0; it < 4; ++it) {
          int gr = bm0 + rF + it * 32;
          v4f fv = {0.f, 0.f, 0.f, 0.f};
          if (!GUARD || gr < M) fv = *(const v4f*)(Af + (long long)gr * lda + k0 + cF);
          fa[it] = fv;
        }
      } else {
#pragma unroll
        for (int it = 0; it < 2; ++it) {
          int gr = bm0 + rT + it * 64;
          v4u dv = {0u, 0u, 0u, 0u};
          if (!GUARD || gr < M) dv = *(const v4u*)(Ab + (long long)gr * lda + k0 + cT);
          ua[it] = dv;
        }
      }
#pragma unroll
      for (int it = 0; it < 2; ++it) {
        int gr = bn0 + rT + it * 64;
        v4u dv = {0u, 0u, 0u, 0u};
        if (!GUARD || gr < N) dv = *(const v4u*)(Bp + (long long)gr * ldb + k0 + cT);
        ub[it] = dv;
      }
    };
    auto storeRegs = [&](int buf) {
      if (AF32) {
#pragma unroll
        for (int it = 0; it < 4; ++it) {
          v4us sv;
          sv[0] = f32_to_bf16(fa[it][0]); sv[1] = f32_to_bf16(fa[it][1]);
          sv[2] = f32_to_bf16(fa[it][2]); sv[3] = f32_to_bf16(fa[it][3]);
          *(v4us*)(&sA[buf][(rF + it * 32) * 40 + cF]) = sv;
        }
      } else {
#pragma unroll
        for (int it = 0; it < 2; ++it)
          *(v4u*)(&sA[buf][(rT + it * 64) * 40 + cT]) = ua[it];
      }
#pragma unroll
      for (int it = 0; it < 2; ++it)
        *(v4u*)(&sB[buf][(rT + it * 64) * 40 + cT]) = ub[it];
    };
    loadRegs(0);
    storeRegs(0);
    for (int ki = 0; ki < nk; ++ki) {
      const int cur = ki & 1;
      __syncthreads();                       // buf[cur] ready; prior readers of buf[cur^1] done
      if (ki + 1 < nk) loadRegs((ki + 1) << 5);
      wmma_tiles(sA[cur], sB[cur], wm, wn, hi, ln, acc);
      if (ki + 1 < nk) storeRegs(cur ^ 1);
    }
  }

  // ---- epilogue: C/D layout col = l%16, row = r + hi*8 ----
#pragma unroll
  for (int tm = 0; tm < 4; ++tm) {
#pragma unroll
    for (int tn = 0; tn < 2; ++tn) {
      int col = bn0 + wn * 32 + tn * 16 + ln;
      if (GUARD && col >= N) continue;
#pragma unroll
      for (int r = 0; r < 8; ++r) {
        int row = bm0 + wm * 64 + tm * 16 + hi * 8 + r;
        if (GUARD && row >= M) continue;
        float v = acc[tm][tn][r];
        long long idx = cOff + (long long)row * ldc + col;
        if (EPI == 0) {
          ((unsigned short*)C)[idx] = f32_to_bf16(v);
        } else if (EPI == 1) {
          ((unsigned short*)C)[idx] = f32_to_bf16(gelu_f(v + bias[col]));
        } else if (EPI == 2) {
          ((float*)C)[idx] = v * scale;
        } else {
          ((float*)C)[idx] = res[idx] + v + bias[col];
        }
      }
    }
  }
}

// ---------------------------------------------------------------------------
// LayerNorm over last dim (512) -> bf16; one block per token
// ---------------------------------------------------------------------------
__global__ __launch_bounds__(256) void ln_kernel(const float* __restrict__ x,
                                                 const float* __restrict__ g,
                                                 const float* __restrict__ bb,
                                                 unsigned short* __restrict__ out) {
  const int t = blockIdx.x, tid = threadIdx.x;
  const float* p = x + (long long)t * DIM;
  float v0 = p[tid], v1 = p[tid + 256];
  __shared__ float red[256];
  red[tid] = v0 + v1;
  __syncthreads();
  for (int o = 128; o; o >>= 1) { if (tid < o) red[tid] += red[tid + o]; __syncthreads(); }
  float mu = red[0] * (1.f / DIM);
  __syncthreads();
  float d0 = v0 - mu, d1 = v1 - mu;
  red[tid] = d0 * d0 + d1 * d1;
  __syncthreads();
  for (int o = 128; o; o >>= 1) { if (tid < o) red[tid] += red[tid + o]; __syncthreads(); }
  float rstd = rsqrtf(red[0] * (1.f / DIM) + 1e-5f);
  unsigned short* q = out + (long long)t * DIM;
  q[tid] = f32_to_bf16(d0 * rstd * g[tid] + bb[tid]);
  q[tid + 256] = f32_to_bf16(d1 * rstd * g[tid + 256] + bb[tid + 256]);
}

// ---------------------------------------------------------------------------
// Masked softmax in place over rows of one layer's attention slab.
// row = (b*H + h)*SEQ + q ; column availability: col<4*512 -> mask[b][col>>9]
// ---------------------------------------------------------------------------
__global__ __launch_bounds__(256) void softmax_kernel(float* __restrict__ attn,
                                                      const int* __restrict__ mask) {
  const int row = blockIdx.x;
  const int b = row / (HEADS * SEQ);
  float* p = attn + (long long)row * SEQ;
  const int* mrow = mask + b * 4;
  __shared__ float red[256];
  const int tid = threadIdx.x;

  float m = -3.0e38f;
  for (int i = tid; i < SEQ; i += 256) {
    bool av = (i >= 2048) || (mrow[i >> 9] != 0);
    if (av) m = fmaxf(m, p[i]);
  }
  red[tid] = m;
  __syncthreads();
  for (int o = 128; o; o >>= 1) { if (tid < o) red[tid] = fmaxf(red[tid], red[tid + o]); __syncthreads(); }
  m = red[0];
  __syncthreads();

  float s = 0.f;
  for (int i = tid; i < SEQ; i += 256) {
    bool av = (i >= 2048) || (mrow[i >> 9] != 0);
    float e = av ? __expf(p[i] - m) : 0.f;
    p[i] = e;
    s += e;
  }
  red[tid] = s;
  __syncthreads();
  for (int o = 128; o; o >>= 1) { if (tid < o) red[tid] += red[tid + o]; __syncthreads(); }
  float inv = 1.f / red[0];
  for (int i = tid; i < SEQ; i += 256) p[i] *= inv;
}

// vt[b][h][d][n] = qkv[(b*SEQ+n)*1536 + 1024 + h*64 + d]
__global__ __launch_bounds__(256) void vt_kernel(const unsigned short* __restrict__ qkv,
                                                 unsigned short* __restrict__ vt) {
  long long idx = (long long)blockIdx.x * 256 + threadIdx.x;
  const long long total = (long long)BATCH * HEADS * HD * SEQ;
  if (idx >= total) return;
  int n = (int)(idx % SEQ);
  long long r = idx / SEQ;
  int d = (int)(r % HD); r /= HD;
  int h = (int)(r % HEADS);
  int b = (int)(r / HEADS);
  vt[idx] = qkv[((long long)(b * SEQ + n)) * (3 * DIM) + 1024 + h * HD + d];
}

// WT[d][n][k] = bf16( W[d][k][n] )   for stacked weights [DEPTH,K,N]
__global__ __launch_bounds__(256) void wcvt_kernel(const float* __restrict__ W,
                                                   unsigned short* __restrict__ WT,
                                                   int K, int N, long long total) {
  long long idx = (long long)blockIdx.x * 256 + threadIdx.x;
  if (idx >= total) return;
  int k = (int)(idx % K);
  long long r = idx / K;
  int n = (int)(r % N);
  long long d = r / N;
  WT[idx] = f32_to_bf16(W[d * (long long)K * N + (long long)k * N + n]);
}

// ---------------------------------------------------------------------------
extern "C" void kernel_launch(void* const* d_in, const int* in_sizes, int n_in,
                              void* d_out, int out_size, void* d_ws, size_t ws_size,
                              hipStream_t stream) {
  const float* x_in   = (const float*)d_in[0];
  const int*   mask   = (const int*)d_in[1];
  const float* ln1_g  = (const float*)d_in[2];
  const float* ln1_b  = (const float*)d_in[3];
  const float* qkv_w  = (const float*)d_in[4];
  const float* proj_w = (const float*)d_in[5];
  const float* proj_b = (const float*)d_in[6];
  const float* ln2_g  = (const float*)d_in[7];
  const float* ln2_b  = (const float*)d_in[8];
  const float* ffn_w1 = (const float*)d_in[9];
  const float* ffn_b1 = (const float*)d_in[10];
  const float* ffn_w2 = (const float*)d_in[11];
  const float* ffn_b2 = (const float*)d_in[12];
  (void)in_sizes; (void)n_in; (void)out_size; (void)ws_size;

  char* ws = (char*)d_ws;
  // workspace layout (bytes)
  float*          x     = (float*)(ws + 0);                       // 5120*512 f32
  unsigned short* h     = (unsigned short*)(ws + 10485760);       // 5120*512 bf16
  unsigned short* qkv   = (unsigned short*)(ws + 15728640);       // 5120*1536 bf16
  unsigned short* vt    = (unsigned short*)(ws + 31457280);       // 16*64*2560 bf16
  unsigned short* obuf  = (unsigned short*)(ws + 36700160);       // 5120*512 bf16
  unsigned short* u     = (unsigned short*)(ws + 41943040);       // 5120*4096 bf16
  unsigned short* qkvT  = (unsigned short*)(ws + 83886080);       // 3*1536*512 bf16
  unsigned short* projT = (unsigned short*)(ws + 88604672);       // 3*512*512 bf16
  unsigned short* w1T   = (unsigned short*)(ws + 90177536);       // 3*4096*512 bf16
  unsigned short* w2T   = (unsigned short*)(ws + 102760448);      // 3*512*4096 bf16

  float* outx = (float*)d_out;
  float* attnBase = outx + (long long)BATCH * SEQ * DIM;          // + 2621440

  // ---- weight convert+transpose (bf16 [out,in]) ----
  {
    long long t1 = (long long)DEPTH * DIM * (3 * DIM);
    wcvt_kernel<<<(unsigned)((t1 + 255) / 256), 256, 0, stream>>>(qkv_w, qkvT, DIM, 3 * DIM, t1);
    long long t2 = (long long)DEPTH * DIM * DIM;
    wcvt_kernel<<<(unsigned)((t2 + 255) / 256), 256, 0, stream>>>(proj_w, projT, DIM, DIM, t2);
    long long t3 = (long long)DEPTH * DIM * MLP_DIM;
    wcvt_kernel<<<(unsigned)((t3 + 255) / 256), 256, 0, stream>>>(ffn_w1, w1T, DIM, MLP_DIM, t3);
    wcvt_kernel<<<(unsigned)((t3 + 255) / 256), 256, 0, stream>>>(ffn_w2, w2T, MLP_DIM, DIM, t3);
  }

  (void)hipMemcpyAsync(x, x_in, (size_t)NTOK * DIM * sizeof(float), hipMemcpyDeviceToDevice, stream);

  const long long SB = (long long)SEQ * (3 * DIM);  // per-batch qkv stride
  const long long NN = (long long)SEQ * SEQ;

  for (int j = 0; j < DEPTH; ++j) {
    float* attnL = attnBase + (long long)j * BATCH * HEADS * NN;

    // LN1 -> h
    ln_kernel<<<NTOK, 256, 0, stream>>>(x, ln1_g + j * DIM, ln1_b + j * DIM, h);

    // qkv = h @ qkv_w   (bf16 out)  M=5120 N=1536 K=512
    gemm_k<0, false, false><<<dim3(12, 40, 1), 256, 0, stream>>>(
        h, 0, DIM, 0, 0,
        qkvT + (long long)j * (3 * DIM) * DIM, 0, DIM, 0, 0,
        qkv, 0, 3 * DIM, 0, 0,
        nullptr, nullptr, 1.f, NTOK, 3 * DIM, DIM, 1);

    // logits = scale * Q @ K^T  -> d_out attn slab (f32)  per (b,h)
    gemm_k<2, false, false><<<dim3(20, 20, BATCH * HEADS), 256, 0, stream>>>(
        qkv, 0, 3 * DIM, SB, HD,                       // Q: base 0, +h*64, +b*SB
        qkv, DIM, 3 * DIM, SB, HD,                     // K: base 512
        attnL, 0, SEQ, (long long)HEADS * NN, NN,
        nullptr, nullptr, SCALE_ATTN, SEQ, SEQ, HD, HEADS);

    // masked softmax in place
    softmax_kernel<<<BATCH * HEADS * SEQ, 256, 0, stream>>>(attnL, mask);

    // V transpose repack
    {
      long long tv = (long long)BATCH * HEADS * HD * SEQ;
      vt_kernel<<<(unsigned)((tv + 255) / 256), 256, 0, stream>>>(qkv, vt);
    }

    // O = P @ V  (A is fp32 probs from d_out, converted on stage) -> obuf bf16
    gemm_k<0, true, true><<<dim3(1, 20, BATCH * HEADS), 256, 0, stream>>>(
        attnL, 0, SEQ, (long long)HEADS * NN, NN,
        vt, 0, SEQ, (long long)HEADS * HD * SEQ, (long long)HD * SEQ,
        obuf, 0, DIM, (long long)SEQ * DIM, HD,
        nullptr, nullptr, 1.f, SEQ, HD, SEQ, HEADS);

    // x += O @ proj_w + proj_b   M=5120 N=512 K=512
    gemm_k<3, false, false><<<dim3(4, 40, 1), 256, 0, stream>>>(
        obuf, 0, DIM, 0, 0,
        projT + (long long)j * DIM * DIM, 0, DIM, 0, 0,
        x, 0, DIM, 0, 0,
        proj_b + j * DIM, x, 1.f, NTOK, DIM, DIM, 1);

    // LN2 -> h
    ln_kernel<<<NTOK, 256, 0, stream>>>(x, ln2_g + j * DIM, ln2_b + j * DIM, h);

    // u = gelu(h @ w1 + b1)  M=5120 N=4096 K=512
    gemm_k<1, false, false><<<dim3(32, 40, 1), 256, 0, stream>>>(
        h, 0, DIM, 0, 0,
        w1T + (long long)j * MLP_DIM * DIM, 0, DIM, 0, 0,
        u, 0, MLP_DIM, 0, 0,
        ffn_b1 + j * MLP_DIM, nullptr, 1.f, NTOK, MLP_DIM, DIM, 1);

    // x += u @ w2 + b2  M=5120 N=512 K=4096
    gemm_k<3, false, false><<<dim3(4, 40, 1), 256, 0, stream>>>(
        u, 0, MLP_DIM, 0, 0,
        w2T + (long long)j * DIM * MLP_DIM, 0, MLP_DIM, 0, 0,
        x, 0, DIM, 0, 0,
        ffn_b2 + j * DIM, x, 1.f, NTOK, DIM, MLP_DIM, 1);
  }

  (void)hipMemcpyAsync(outx, x, (size_t)NTOK * DIM * sizeof(float), hipMemcpyDeviceToDevice, stream);
}
